// SelfAttention_33354716021115
// MI455X (gfx1250) — matrix operand order repaired and checked by reading, not compiled, not run on hardware
//
#include <hip/hip_runtime.h>

typedef __attribute__((ext_vector_type(16))) __bf16 v16bf;
typedef __attribute__((ext_vector_type(8)))  __bf16 v8bf;
typedef __attribute__((ext_vector_type(8)))  float  v8f;
typedef unsigned short u16;
typedef unsigned int   u32;

// ---------- helpers ----------
__device__ __forceinline__ u16 f2bf(float f) {
    u32 u = __float_as_uint(f);
    u += 0x7FFFu + ((u >> 16) & 1u);   // round-to-nearest-even
    return (u16)(u >> 16);
}

__device__ __forceinline__ v16bf mk16(v8bf lo, v8bf hi) {
    return __builtin_shufflevector(lo, hi, 0,1,2,3,4,5,6,7,8,9,10,11,12,13,14,15);
}

// swap data between the two 16-lane rows (VALU, no LDS): lane j <- other row, lane j
__device__ __forceinline__ float permx16(float v) {
    int r = __builtin_amdgcn_permlanex16(__float_as_int(v), __float_as_int(v),
                                         0x76543210u, 0xfedcba98u, false, false);
    return __int_as_float(r);
}

// ---------- f32 -> bf16 convert ----------
__global__ __launch_bounds__(256)
void convert_f32_bf16(const float* __restrict__ src, u16* __restrict__ dst, int n) {
    int i = (blockIdx.x * 256 + threadIdx.x) * 4;
    if (i >= n) return;
    float4 v = *reinterpret_cast<const float4*>(src + i);
    u32 lo = (u32)f2bf(v.x) | ((u32)f2bf(v.y) << 16);
    u32 hi = (u32)f2bf(v.z) | ((u32)f2bf(v.w) << 16);
    *reinterpret_cast<uint2*>(dst + i) = make_uint2(lo, hi);
}

// ---------- GEMM: C[M,N] = scale * (A[M,K] (bf16) @ W[N,K]^T (bf16)) ----------
// block = 256 threads = 8 waves (4 along M x 2 along N); block tile 128x64;
// per-wave tile 32x32 (2x2 WMMA 16x16); K-step 32; LDS tiles with
// register software pipeline (prefetch tile t+1 while computing tile t).
template <bool BF16OUT>
__global__ __launch_bounds__(256)
void gemm_bf16(const u16* __restrict__ A, const u16* __restrict__ W,
               u16* __restrict__ Cb, float* __restrict__ Cf,
               const float* __restrict__ bias, float scale, int M, int N, int K)
{
    constexpr int AP = 48, BP = 48, CP = 72;   // padded pitches (16B aligned rows)
    __shared__ u16 At[128 * AP];
    __shared__ u16 Bt[64 * BP];
    __shared__ u16 Ct[BF16OUT ? 128 * CP : 1];

    const int tid  = threadIdx.x;
    const int wave = tid >> 5;
    const int lane = tid & 31;
    const int ln = lane & 15, lh = lane >> 4;
    const int wm = wave >> 1, wn = wave & 1;
    const int m0 = blockIdx.y * 128;
    const int n0 = blockIdx.x * 64;

    // staging assignment (constant per thread)
    const int ar0 = (tid * 2) >> 2,     ac0 = ((tid * 2) & 3) * 8;
    const int ar1 = (tid * 2 + 1) >> 2, ac1 = ((tid * 2 + 1) & 3) * 8;
    const int br0 = tid >> 2,           bc0 = (tid & 3) * 8;
    const u16* Apt = A + (size_t)(m0 + ar0) * K + ac0;
    const u16* Apt1 = A + (size_t)(m0 + ar1) * K + ac1;
    const u16* Wpt = W + (size_t)(n0 + br0) * K + bc0;

    v8f acc[2][2] = {};
    uint4 ra0, ra1, rb;

    // prefetch tile 0
    ra0 = *reinterpret_cast<const uint4*>(Apt);
    ra1 = *reinterpret_cast<const uint4*>(Apt1);
    rb  = *reinterpret_cast<const uint4*>(Wpt);

    for (int kt = 0; kt < K; kt += 32) {
        __syncthreads();
        *reinterpret_cast<uint4*>(&At[ar0 * AP + ac0]) = ra0;
        *reinterpret_cast<uint4*>(&At[ar1 * AP + ac1]) = ra1;
        *reinterpret_cast<uint4*>(&Bt[br0 * BP + bc0]) = rb;
        __syncthreads();

        // prefetch next tile into registers (overlaps with WMMA below)
        if (kt + 32 < K) {
            ra0 = *reinterpret_cast<const uint4*>(Apt  + kt + 32);
            ra1 = *reinterpret_cast<const uint4*>(Apt1 + kt + 32);
            rb  = *reinterpret_cast<const uint4*>(Wpt  + kt + 32);
        }

        // A fragments: 16-bit A layout — lane half lh selects K {0-7,16-23} / {8-15,24-31}
        v16bf afrag[2];
        #pragma unroll
        for (int ms = 0; ms < 2; ++ms) {
            int r = wm * 32 + ms * 16 + ln;
            v8bf lo = *reinterpret_cast<const v8bf*>(&At[r * AP + lh * 8]);
            v8bf hi = *reinterpret_cast<const v8bf*>(&At[r * AP + 16 + lh * 8]);
            afrag[ms] = mk16(lo, hi);
        }
        // B fragments: 16-bit B layout — lane half lh selects K {0-7,16-23} / {8-15,24-31}
        v16bf bfrag[2];
        #pragma unroll
        for (int ns = 0; ns < 2; ++ns) {
            int r = wn * 32 + ns * 16 + ln;
            v8bf lo = *reinterpret_cast<const v8bf*>(&Bt[r * BP + lh * 8]);
            v8bf hi = *reinterpret_cast<const v8bf*>(&Bt[r * BP + 16 + lh * 8]);
            bfrag[ns] = mk16(lo, hi);
        }
        #pragma unroll
        for (int ms = 0; ms < 2; ++ms)
            #pragma unroll
            for (int ns = 0; ns < 2; ++ns)
                acc[ms][ns] = __builtin_amdgcn_wmma_f32_16x16x32_bf16(
                    false, afrag[ms], false, bfrag[ns], (short)0, acc[ms][ns], false, false);
    }

    // C/D layout — lane l, vgpr r -> m = r + 8*(l>>4), n = l&15
    if constexpr (BF16OUT) {
        __syncthreads();
        #pragma unroll
        for (int ms = 0; ms < 2; ++ms)
            #pragma unroll
            for (int ns = 0; ns < 2; ++ns)
                #pragma unroll
                for (int r = 0; r < 8; ++r)
                    Ct[(wm * 32 + ms * 16 + r + 8 * lh) * CP + wn * 32 + ns * 16 + ln] =
                        f2bf(acc[ms][ns][r] * scale);
        __syncthreads();
        // rows 0-63 x 64 cols = 512 x 16B chunks, 2 per thread
        #pragma unroll
        for (int i = 0; i < 2; ++i) {
            int cid = tid * 2 + i;
            int row = cid >> 3, col = (cid & 7) * 8;
            uint4 v = *reinterpret_cast<const uint4*>(&Ct[row * CP + col]);
            *reinterpret_cast<uint4*>(Cb + (size_t)(m0 + row) * N + n0 + col) = v;
        }
    } else {
        float* cbase = Cf + (size_t)m0 * N + n0;
        #pragma unroll
        for (int ms = 0; ms < 2; ++ms)
            #pragma unroll
            for (int ns = 0; ns < 2; ++ns) {
                int n = wn * 32 + ns * 16 + ln;
                float bv = bias[n0 + n];
                #pragma unroll
                for (int r = 0; r < 8; ++r) {
                    int m = wm * 32 + ms * 16 + r + 8 * lh;
                    cbase[(size_t)m * N + n] = acc[ms][ns][r] + bv;
                }
            }
    }
}

// ---------- flash attention: per (batch, head, 64-row Q tile) ----------
// 128 threads = 4 waves; wave w owns Q rows [q0+16w, q0+16w+16).
// Register-pipelined K/V staging; V transposed via packed dword scatter;
// QK^T (4 WMMA, all K fragments preloaded); softmax in VALU with one
// v_permlanex16 per reduction; P assembled in registers; PV (4 WMMA).
#define S_LEN 2048
__global__ __launch_bounds__(128)
void attn_kernel(const u16* __restrict__ Qp, const u16* __restrict__ Kp,
                 const u16* __restrict__ Vp, u16* __restrict__ Op)
{
    constexpr int KP = 80, VP = 48, SP = 36, OP = 72;
    __shared__ u16   Kt[32 * KP];        // [key][d]
    __shared__ u16   Vt[64 * VP];        // [d][key]  (transposed)
    __shared__ float Sb[4 * 16 * SP];    // per-wave raw score tile [m][key]
    __shared__ u16   Ot[64 * OP];        // output bounce for coalesced stores

    const int tid  = threadIdx.x;
    const int wave = tid >> 5;
    const int lane = tid & 31;
    const int ln = lane & 15, lh = lane >> 4;

    int bx = blockIdx.x;
    const int qt = bx & 31;  bx >>= 5;   // S/64 = 32 tiles
    const int h  = bx & 15;  bx >>= 4;   // 16 heads
    const int b  = bx;                   // batch
    const int q0 = qt * 64;
    const size_t bbase = (size_t)b * S_LEN * 1024 + (size_t)h * 64;

    // staging assignment (constant per thread)
    const int krow0 = tid >> 3, kcol = (tid & 7) * 8;   // K rows tid>>3 and +16
    const int vk0 = (tid >> 3) * 2;                     // V key pair, same col
    const u16* Kpt0 = Kp + bbase + (size_t)krow0 * 1024 + kcol;
    const u16* Kpt1 = Kp + bbase + (size_t)(krow0 + 16) * 1024 + kcol;
    const u16* Vpt0 = Vp + bbase + (size_t)vk0 * 1024 + kcol;
    const u16* Vpt1 = Vp + bbase + (size_t)(vk0 + 1) * 1024 + kcol;

    // loop-invariant Q fragments (16 x 64, pre-scaled by 1/sqrt(d) upstream)
    v16bf qf[2];
    {
        const size_t qrow = bbase + (size_t)(q0 + wave * 16 + ln) * 1024;
        #pragma unroll
        for (int kk = 0; kk < 2; ++kk) {
            v8bf lo = *reinterpret_cast<const v8bf*>(Qp + qrow + kk * 32 + lh * 8);
            v8bf hi = *reinterpret_cast<const v8bf*>(Qp + qrow + kk * 32 + 16 + lh * 8);
            qf[kk] = mk16(lo, hi);
        }
    }

    float mrow = -3.0e38f, lrow = 0.f;   // per-lane: stats of row ln
    v8f o[4] = {};

    float* Sw = &Sb[wave * 16 * SP];

    // prefetch tile 0
    uint4 rk0 = *reinterpret_cast<const uint4*>(Kpt0);
    uint4 rk1 = *reinterpret_cast<const uint4*>(Kpt1);
    uint4 rv0 = *reinterpret_cast<const uint4*>(Vpt0);
    uint4 rv1 = *reinterpret_cast<const uint4*>(Vpt1);

    for (int kb = 0; kb < S_LEN; kb += 32) {
        __syncthreads();
        // K tile row-major (b128 stores)
        *reinterpret_cast<uint4*>(&Kt[krow0 * KP + kcol])        = rk0;
        *reinterpret_cast<uint4*>(&Kt[(krow0 + 16) * KP + kcol]) = rk1;
        // V tile transposed: pack (d, key even|odd) pairs -> dword scatter
        {
            const u16* a16 = reinterpret_cast<const u16*>(&rv0);
            const u16* b16 = reinterpret_cast<const u16*>(&rv1);
            #pragma unroll
            for (int e = 0; e < 8; ++e) {
                u32 pk = (u32)a16[e] | ((u32)b16[e] << 16);
                *reinterpret_cast<u32*>(&Vt[(kcol + e) * VP + vk0]) = pk;
            }
        }
        __syncthreads();

        // prefetch next tile into registers (overlaps with WMMA/softmax below)
        if (kb + 32 < S_LEN) {
            rk0 = *reinterpret_cast<const uint4*>(Kpt0 + (size_t)(kb + 32) * 1024);
            rk1 = *reinterpret_cast<const uint4*>(Kpt1 + (size_t)(kb + 32) * 1024);
            rv0 = *reinterpret_cast<const uint4*>(Vpt0 + (size_t)(kb + 32) * 1024);
            rv1 = *reinterpret_cast<const uint4*>(Vpt1 + (size_t)(kb + 32) * 1024);
        }

        // preload all 4 K fragments (distinct registers -> pipelined ds_loads)
        v16bf kf[2][2];
        #pragma unroll
        for (int g = 0; g < 2; ++g)
            #pragma unroll
            for (int kk = 0; kk < 2; ++kk) {
                const u16* kp = &Kt[(g * 16 + ln) * KP + kk * 32 + lh * 8];
                kf[g][kk] = mk16(*reinterpret_cast<const v8bf*>(kp),
                                 *reinterpret_cast<const v8bf*>(kp + 16));
            }

        // scores S = Q K^T  (two independent accumulator chains)
        v8f s[2];
        #pragma unroll
        for (int g = 0; g < 2; ++g) {
            v8f z = __builtin_amdgcn_wmma_f32_16x16x32_bf16(
                        false, qf[0], false, kf[g][0], (short)0, v8f{}, false, false);
            s[g] = __builtin_amdgcn_wmma_f32_16x16x32_bf16(
                        false, qf[1], false, kf[g][1], (short)0, z, false, false);
        }

        // bounce scores: C-layout scatter -> each lane reads its row in
        // A-fragment element order (row ln; keys lh*8..+7 and 16+lh*8..+7)
        #pragma unroll
        for (int r = 0; r < 8; ++r) {
            int m = r + 8 * lh;
            Sw[m * SP + ln]      = s[0][r];
            Sw[m * SP + 16 + ln] = s[1][r];
        }
        float sc[16];
        {
            const float* sp = &Sw[ln * SP + lh * 8];
            float4 c0 = *reinterpret_cast<const float4*>(sp);
            float4 c1 = *reinterpret_cast<const float4*>(sp + 4);
            float4 c2 = *reinterpret_cast<const float4*>(sp + 16);
            float4 c3 = *reinterpret_cast<const float4*>(sp + 20);
            sc[0]=c0.x; sc[1]=c0.y; sc[2]=c0.z; sc[3]=c0.w;
            sc[4]=c1.x; sc[5]=c1.y; sc[6]=c1.z; sc[7]=c1.w;
            sc[8]=c2.x; sc[9]=c2.y; sc[10]=c2.z; sc[11]=c2.w;
            sc[12]=c3.x; sc[13]=c3.y; sc[14]=c3.z; sc[15]=c3.w;
        }

        // row softmax: local VALU + one permlanex16 per reduction
        float mx = sc[0];
        #pragma unroll
        for (int i = 1; i < 16; ++i) mx = fmaxf(mx, sc[i]);
        mx = fmaxf(mx, permx16(mx));
        float mn = fmaxf(mrow, mx);
        float alpha = __expf(mrow - mn);
        mrow = mn;

        u16 pb[16];
        float psum = 0.f;
        #pragma unroll
        for (int i = 0; i < 16; ++i) {
            float p = __expf(sc[i] - mn);
            psum += p;
            pb[i] = f2bf(p);
        }
        psum += permx16(psum);
        lrow = lrow * alpha + psum;

        // P fragment directly in registers (already in A-fragment order)
        union { u16 us[16]; v16bf v; } pu;
        #pragma unroll
        for (int i = 0; i < 16; ++i) pu.us[i] = pb[i];
        v16bf pf = pu.v;

        // broadcast alpha into C-layout rows (8 bpermutes) and rescale O
        #pragma unroll
        for (int r = 0; r < 8; ++r) {
            float av = __shfl(alpha, r + 8 * lh, 32);
            #pragma unroll
            for (int hg = 0; hg < 4; ++hg) o[hg][r] *= av;
        }

        // O += P (16x32) x V (32x64): V^T rows are contiguous in Vt
        v16bf vf[4];
        #pragma unroll
        for (int hg = 0; hg < 4; ++hg) {
            const u16* vp = &Vt[(hg * 16 + ln) * VP + lh * 8];
            vf[hg] = mk16(*reinterpret_cast<const v8bf*>(vp),
                          *reinterpret_cast<const v8bf*>(vp + 16));
        }
        #pragma unroll
        for (int hg = 0; hg < 4; ++hg)
            o[hg] = __builtin_amdgcn_wmma_f32_16x16x32_bf16(
                        false, pf, false, vf[hg], (short)0, o[hg], false, false);
    }

    // normalize, bounce through LDS, coalesced b128 stores
    float linv = 1.0f / lrow;
    __syncthreads();
    #pragma unroll
    for (int r = 0; r < 8; ++r) {
        float lc = __shfl(linv, r + 8 * lh, 32);
        #pragma unroll
        for (int hg = 0; hg < 4; ++hg)
            Ot[(wave * 16 + r + 8 * lh) * OP + hg * 16 + ln] = f2bf(o[hg][r] * lc);
    }
    __syncthreads();
    // 64 rows x 64 cols = 512 x 16B chunks, 4 per thread
    #pragma unroll
    for (int i = 0; i < 4; ++i) {
        int cid = tid + 128 * i;
        int row = cid >> 3, col = (cid & 7) * 8;
        uint4 v = *reinterpret_cast<const uint4*>(&Ot[row * OP + col]);
        *reinterpret_cast<uint4*>(Op + bbase + (size_t)(q0 + row) * 1024 + col) = v;
    }
}

// ---------- host ----------
extern "C" void kernel_launch(void* const* d_in, const int* in_sizes, int n_in,
                              void* d_out, int out_size, void* d_ws, size_t ws_size,
                              hipStream_t stream)
{
    const float* values = (const float*)d_in[0];
    const float* keysf  = (const float*)d_in[1];
    const float* queryf = (const float*)d_in[2];
    const float* Wv = (const float*)d_in[3];
    const float* Wk = (const float*)d_in[4];
    const float* Wq = (const float*)d_in[5];
    const float* Wo = (const float*)d_in[6];
    const float* bo = (const float*)d_in[7];
    float* out = (float*)d_out;

    const int B = 2, S = 2048, E = 1024;
    const size_t NA = (size_t)B * S * E;   // 4M activation elements
    const size_t NW = (size_t)E * E;       // 1M weight elements

    char* ws = (char*)d_ws;
    size_t off = 0;
    auto alloc = [&](size_t bytes) -> u16* {
        u16* p = (u16*)(ws + off);
        off += (bytes + 255) & ~(size_t)255;
        return p;
    };
    u16* vals_b = alloc(NA * 2);
    u16* keys_b = alloc(NA * 2);
    u16* qry_b  = alloc(NA * 2);
    u16* Wv_b   = alloc(NW * 2);
    u16* Wk_b   = alloc(NW * 2);
    u16* Wq_b   = alloc(NW * 2);
    u16* Wo_b   = alloc(NW * 2);
    u16* Vp     = alloc(NA * 2);
    u16* Kp     = alloc(NA * 2);
    u16* Qp     = alloc(NA * 2);
    u16* Ao     = alloc(NA * 2);

    const int M = B * S;
    // f32 -> bf16
    convert_f32_bf16<<<(int)(NA / 4 / 256), 256, 0, stream>>>(values, vals_b, (int)NA);
    convert_f32_bf16<<<(int)(NA / 4 / 256), 256, 0, stream>>>(keysf,  keys_b, (int)NA);
    convert_f32_bf16<<<(int)(NA / 4 / 256), 256, 0, stream>>>(queryf, qry_b,  (int)NA);
    convert_f32_bf16<<<(int)(NW / 4 / 256), 256, 0, stream>>>(Wv, Wv_b, (int)NW);
    convert_f32_bf16<<<(int)(NW / 4 / 256), 256, 0, stream>>>(Wk, Wk_b, (int)NW);
    convert_f32_bf16<<<(int)(NW / 4 / 256), 256, 0, stream>>>(Wq, Wq_b, (int)NW);
    convert_f32_bf16<<<(int)(NW / 4 / 256), 256, 0, stream>>>(Wo, Wo_b, (int)NW);

    // Q/K/V projections (x @ W^T); Q pre-scaled by 1/sqrt(HEAD_DIM)
    dim3 gg(E / 64, M / 128);
    gemm_bf16<true><<<gg, 256, 0, stream>>>(qry_b,  Wq_b, Qp, nullptr, nullptr, 0.125f, M, E, E);
    gemm_bf16<true><<<gg, 256, 0, stream>>>(keys_b, Wk_b, Kp, nullptr, nullptr, 1.0f,   M, E, E);
    gemm_bf16<true><<<gg, 256, 0, stream>>>(vals_b, Wv_b, Vp, nullptr, nullptr, 1.0f,   M, E, E);

    // flash attention: B * HEADS * (S/64) blocks
    attn_kernel<<<B * 16 * (S / 64), 128, 0, stream>>>(Qp, Kp, Vp, Ao);

    // output projection + bias -> fp32 out
    gemm_bf16<false><<<gg, 256, 0, stream>>>(Ao, Wo_b, nullptr, out, bo, 1.0f, M, E, E);
}
